// CharRNN_71820443124056
// MI455X (gfx1250) — matrix operand order repaired and checked by reading, not compiled, not run on hardware
//
#include <hip/hip_runtime.h>
#include <stdint.h>

typedef __bf16 bf16_t;
typedef bf16_t v16bf __attribute__((ext_vector_type(16)));
typedef float  v8f   __attribute__((ext_vector_type(8)));

#define T_STEPS 2048
#define BATCH   128
#define WIN     128
#define HID     512
#define VOCAB   128

// bf16 workspace layout (element offsets)
#define OFF_WIH1 0
#define OFF_WHH1 (OFF_WIH1 + HID*WIN)
#define OFF_WIH2 (OFF_WHH1 + HID*HID)
#define OFF_WHH2 (OFF_WIH2 + HID*HID)
#define OFF_WOUT (OFF_WHH2 + HID*HID)
#define BF16_TOTAL (OFF_WOUT + VOCAB*HID)   // 917504 bf16 elements

__device__ __forceinline__ bf16_t f2bf(float f) {
  union { float f; uint32_t u; } x; x.f = f;
  uint32_t r = x.u + 0x7FFFu + ((x.u >> 16) & 1u);   // round-to-nearest-even
  union { uint16_t s; bf16_t b; } y; y.s = (uint16_t)(r >> 16);
  return y.b;
}

union FragU { v16bf v; uint4 u[2]; };

// A-matrix (16x32, M x K) fragment: lane holds row M=lane&15,
// K = {base..base+7, base+16..base+23}, base = (lane&16)?8:0  (ISA 7.12.2)
__device__ __forceinline__ v16bf load_a(const bf16_t* base, int ld, int k0, int lane) {
  const int m  = lane & 15;
  const int kb = k0 + ((lane & 16) ? 8 : 0);
  const bf16_t* p = base + m * ld + kb;
  FragU f;
  f.u[0] = *(const uint4*)(p);
  f.u[1] = *(const uint4*)(p + 16);
  return f.v;
}

// B-matrix (32x16, K x N) fragment built from W stored row-major (N rows, K cols):
// lane holds column N=lane&15, K = {kb..kb+7, kb+16..kb+23}, kb = (lane&16)?8:0
__device__ __forceinline__ v16bf load_b(const bf16_t* wrow, int ld, int k0, int lane) {
  const int n  = lane & 15;
  const int kb = k0 + ((lane & 16) ? 8 : 0);
  const bf16_t* p = wrow + n * ld + kb;
  FragU f;
  f.u[0] = *(const uint4*)(p);
  f.u[1] = *(const uint4*)(p + 16);
  return f.v;
}

#define WMMA_BF16(A, B, C) \
  __builtin_amdgcn_wmma_f32_16x16x32_bf16(false, (A), false, (B), (short)0, (C), false, false)

// ---------------- Phase 0: convert weights to bf16, fold biases ----------------
__global__ void convert_weights(
    const float* __restrict__ W_ih1, const float* __restrict__ W_hh1,
    const float* __restrict__ W_ih2, const float* __restrict__ W_hh2,
    const float* __restrict__ W_out,
    const float* __restrict__ b_ih1, const float* __restrict__ b_hh1,
    const float* __restrict__ b_ih2, const float* __restrict__ b_hh2,
    bf16_t* __restrict__ wbf, float* __restrict__ biasf)
{
  int i = blockIdx.x * blockDim.x + threadIdx.x;
  if (i < BF16_TOTAL) {
    float v;
    if (i < OFF_WHH1)      v = W_ih1[i - OFF_WIH1];
    else if (i < OFF_WIH2) v = W_hh1[i - OFF_WHH1];
    else if (i < OFF_WHH2) v = W_ih2[i - OFF_WIH2];
    else if (i < OFF_WOUT) v = W_hh2[i - OFF_WHH2];
    else                   v = W_out[i - OFF_WOUT];
    wbf[i] = f2bf(v);
  }
  if (i < HID)            biasf[i] = b_ih1[i] + b_hh1[i];
  else if (i < 2 * HID)   biasf[i] = b_ih2[i - HID] + b_hh2[i - HID];
}

// ---------------- Persistent RNN kernel: one workgroup per 16 batch rows -------
__global__ void __launch_bounds__(512) rnn_persistent(
    const float*  __restrict__ dataset,
    const bf16_t* __restrict__ wbf,
    const float*  __restrict__ biasf,
    const float*  __restrict__ b_out,
    float*        __restrict__ out)
{
  __shared__ __align__(16) bf16_t xbuf[16 * WIN];       //  4 KB
  __shared__ __align__(16) bf16_t h1s[16 * HID];        // 16 KB (single buffer)
  __shared__ __align__(16) bf16_t h2s[2][16 * HID];     // 32 KB (double buffer)

  const int tid  = threadIdx.x;
  const int lane = tid & 31;
  const int wave = tid >> 5;          // 16 waves
  const int m0   = blockIdx.x << 4;   // batch-row base for this workgroup
  const int n0   = wave << 5;         // 32-wide H chunk owned by this wave
  const int row  = lane & 15;
  const int mhi  = (lane & 16) ? 8 : 0;

  const bf16_t* wih1 = wbf + OFF_WIH1;
  const bf16_t* whh1 = wbf + OFF_WHH1;
  const bf16_t* wih2 = wbf + OFF_WIH2;
  const bf16_t* whh2 = wbf + OFF_WHH2;
  const bf16_t* wout = wbf + OFF_WOUT;

  for (int i = tid; i < 16 * HID; i += 512) {
    h1s[i]    = f2bf(0.0f);
    h2s[0][i] = f2bf(0.0f);
    h2s[1][i] = f2bf(0.0f);
  }

  const float bias1a = biasf[n0 + row];
  const float bias1b = biasf[n0 + 16 + row];
  const float bias2a = biasf[HID + n0 + row];
  const float bias2b = biasf[HID + n0 + 16 + row];

  __syncthreads();

  const v8f vzero = {0.f, 0.f, 0.f, 0.f, 0.f, 0.f, 0.f, 0.f};
  int cur = 0;

  for (int t = 0; t < T_STEPS; ++t) {
    // stage x_t (fp32 -> bf16) into LDS; prefetch x_{t+1}
    const float* xg = dataset + ((size_t)t * BATCH + m0) * WIN;
#pragma unroll
    for (int i = 0; i < 4; ++i) {
      int idx = tid + i * 512;
      xbuf[idx] = f2bf(xg[idx]);
    }
    if (t + 1 < T_STEPS)
      __builtin_prefetch(dataset + ((size_t)(t + 1) * BATCH + m0) * WIN + tid * 4, 0, 0);
    __syncthreads();                               // xbuf ready

    // ---- layer 1: h1' = tanh(x W_ih1^T + h1 W_hh1^T + bias1) ----
    v8f acc0 = vzero, acc1 = vzero;
#pragma unroll
    for (int kk = 0; kk < WIN; kk += 32) {
      v16bf a  = load_a(xbuf, WIN, kk, lane);
      v16bf b0 = load_b(wih1 + n0 * WIN,        WIN, kk, lane);
      v16bf b1 = load_b(wih1 + (n0 + 16) * WIN, WIN, kk, lane);
      acc0 = WMMA_BF16(a, b0, acc0);
      acc1 = WMMA_BF16(a, b1, acc1);
    }
#pragma unroll 4
    for (int kk = 0; kk < HID; kk += 32) {
      v16bf a  = load_a(h1s, HID, kk, lane);
      v16bf b0 = load_b(whh1 + n0 * HID,        HID, kk, lane);
      v16bf b1 = load_b(whh1 + (n0 + 16) * HID, HID, kk, lane);
      acc0 = WMMA_BF16(a, b0, acc0);
      acc1 = WMMA_BF16(a, b1, acc1);
    }
    __syncthreads();                               // all waves done reading old h1
#pragma unroll
    for (int r = 0; r < 8; ++r) {
      int m = r + mhi;
      h1s[m * HID + n0 + row]      = f2bf(tanhf(acc0[r] + bias1a));
      h1s[m * HID + n0 + 16 + row] = f2bf(tanhf(acc1[r] + bias1b));
    }
    __syncthreads();                               // new h1 visible

    // ---- layer 2: h2' = tanh(h1' W_ih2^T + h2 W_hh2^T + bias2) ----
    v8f acc2 = vzero, acc3 = vzero;
#pragma unroll 4
    for (int kk = 0; kk < HID; kk += 32) {
      v16bf a  = load_a(h1s, HID, kk, lane);
      v16bf b0 = load_b(wih2 + n0 * HID,        HID, kk, lane);
      v16bf b1 = load_b(wih2 + (n0 + 16) * HID, HID, kk, lane);
      acc2 = WMMA_BF16(a, b0, acc2);
      acc3 = WMMA_BF16(a, b1, acc3);
    }
    const bf16_t* h2cur = h2s[cur];
#pragma unroll 4
    for (int kk = 0; kk < HID; kk += 32) {
      v16bf a  = load_a(h2cur, HID, kk, lane);
      v16bf b0 = load_b(whh2 + n0 * HID,        HID, kk, lane);
      v16bf b1 = load_b(whh2 + (n0 + 16) * HID, HID, kk, lane);
      acc2 = WMMA_BF16(a, b0, acc2);
      acc3 = WMMA_BF16(a, b1, acc3);
    }
    bf16_t* h2nxt = h2s[cur ^ 1];                  // double-buffered: no barrier needed
#pragma unroll
    for (int r = 0; r < 8; ++r) {
      int m = r + mhi;
      h2nxt[m * HID + n0 + row]      = f2bf(tanhf(acc2[r] + bias2a));
      h2nxt[m * HID + n0 + 16 + row] = f2bf(tanhf(acc3[r] + bias2b));
    }
    cur ^= 1;
    __syncthreads();                               // new h2 visible; h1/h2 reads retired
  }

  // ---- head: out = h2_final @ W_out^T + b_out (waves 0..7, 16x16 tiles) ----
  if (wave < 8) {
    const int v0 = wave << 4;
    const bf16_t* h2f = h2s[cur];
    v8f acc = vzero;
#pragma unroll 4
    for (int kk = 0; kk < HID; kk += 32) {
      v16bf a = load_a(h2f, HID, kk, lane);
      v16bf b = load_b(wout + v0 * HID, HID, kk, lane);
      acc = WMMA_BF16(a, b, acc);
    }
    const float bo = b_out[v0 + row];
#pragma unroll
    for (int r = 0; r < 8; ++r) {
      int m = r + mhi;
      out[(size_t)(m0 + m) * VOCAB + v0 + row] = acc[r] + bo;
    }
  }
}

extern "C" void kernel_launch(void* const* d_in, const int* in_sizes, int n_in,
                              void* d_out, int out_size, void* d_ws, size_t ws_size,
                              hipStream_t stream) {
  const float* dataset = (const float*)d_in[0];
  const float* W_ih1   = (const float*)d_in[1];
  const float* b_ih1   = (const float*)d_in[2];
  const float* W_hh1   = (const float*)d_in[3];
  const float* b_hh1   = (const float*)d_in[4];
  const float* W_ih2   = (const float*)d_in[5];
  const float* b_ih2   = (const float*)d_in[6];
  const float* W_hh2   = (const float*)d_in[7];
  const float* b_hh2   = (const float*)d_in[8];
  const float* W_out   = (const float*)d_in[9];
  const float* b_out   = (const float*)d_in[10];

  bf16_t* wbf  = (bf16_t*)d_ws;
  float* biasf = (float*)((char*)d_ws + (size_t)BF16_TOTAL * sizeof(bf16_t));

  int cvt_blocks = (BF16_TOTAL + 255) / 256;
  convert_weights<<<cvt_blocks, 256, 0, stream>>>(
      W_ih1, W_hh1, W_ih2, W_hh2, W_out,
      b_ih1, b_hh1, b_ih2, b_hh2, wbf, biasf);

  rnn_persistent<<<BATCH / 16, 512, 0, stream>>>(dataset, wbf, biasf, b_out, (float*)d_out);
}
